// VectorQuantizer_47880295416496
// MI455X (gfx1250) — matrix operand order repaired and checked by reading, not compiled, not run on hardware
//
#include <hip/hip_runtime.h>
#include <cstdint>
#include <cstddef>

#define NUM_CENTROIDS 2048
#define DIM 256
#define N_ROWS 32768   // 32*1024 rows
#define EMA_DECAY 0.99f
#define COMMIT 0.25f

#define KB_TILE 32                        // centroids staged per chunk (2 WMMA N-tiles)
#define NKB (NUM_CENTROIDS / KB_TILE)     // 64 chunks
#define CHUNK_HALFS (KB_TILE * DIM)       // 8192 f16 = 16 KB

typedef __attribute__((ext_vector_type(16))) _Float16 v16h;
typedef __attribute__((ext_vector_type(8)))  float    v8f;

// Use the CDNA5 Tensor Data Mover for LDS staging when the builtin exists.
// This toolchain exposes the 6-arg form:
//   (uint32x4 g0, int32x8 g1, int32x4 g2, int32x4 g3, int32x8 extra, i32 cpol)
// Host pass / other toolchains fall back to the sync global->LDS
// copy. LDS layout identical in both paths.
#if defined(__HIP_DEVICE_COMPILE__) && \
    __has_builtin(__builtin_amdgcn_tensor_load_to_lds) && \
    __has_builtin(__builtin_amdgcn_s_wait_tensorcnt)
#define HAVE_TDM 1
#else
#define HAVE_TDM 0
#endif

#if HAVE_TDM
typedef unsigned int u32x4 __attribute__((ext_vector_type(4)));
typedef int          i32x8 __attribute__((ext_vector_type(8)));
typedef int          i32x4 __attribute__((ext_vector_type(4)));

// Issue one TDM DMA: KB_TILE x DIM f16 tile (contiguous rows) -> LDS.
// D# per ISA ch.8: group0 = {count=1 | lds_addr | global_addr[56:0] | type=2},
// group1 = {data_size=2B, tensor_dim0=DIM, tensor_dim1=KB_TILE,
//           tile_dim0=DIM, tile_dim1=KB_TILE, tensor_dim0_stride=DIM}.
__device__ __forceinline__ void tdm_load_chunk(const _Float16* gsrc,
                                               unsigned lds_addr_bytes) {
  unsigned long long ga = (unsigned long long)(size_t)gsrc;
  u32x4 g0;
  g0[0] = 1u;                                             // count=1 (user D#)
  g0[1] = lds_addr_bytes;                                 // lds_addr
  g0[2] = (unsigned)(ga & 0xFFFFFFFFu);                   // global_addr[31:0]
  g0[3] = (unsigned)((ga >> 32) & 0x01FFFFFFu)            // global_addr[56:32]
          | (2u << 30);                                   // type=2 ("image")
  i32x8 g1;
  g1[0] = (1 << 16);                                      // data_size=1 -> 2 bytes
  g1[1] = (DIM & 0xFFFF) << 16;                           // tensor_dim0[15:0]
  g1[2] = (DIM >> 16) | ((KB_TILE & 0xFFFF) << 16);       // tdim0 hi | tensor_dim1 lo
  g1[3] = (KB_TILE >> 16) | ((DIM & 0xFFFF) << 16);       // tdim1 hi | tile_dim0
  g1[4] = KB_TILE;                                        // tile_dim1 (tile_dim2=0)
  g1[5] = DIM;                                            // tensor_dim0_stride[31:0]
  g1[6] = 0;                                              // stride0 hi | stride1 lo
  g1[7] = 0;
  i32x4 z4 = {0, 0, 0, 0};                                // groups 2/3 unused (2D)
  i32x8 z8 = {0, 0, 0, 0, 0, 0, 0, 0};                    // extra group unused
  __builtin_amdgcn_tensor_load_to_lds(g0, g1, z4, z4, z8, 0);
}
#endif

// ---------------------------------------------------------------------------
// Kernel 1: codebook f32 -> f16 (for WMMA), copy codebook_values output,
//           zero the counts accumulator.
// ---------------------------------------------------------------------------
__global__ void vq_prep(const float* __restrict__ codebook,
                        _Float16* __restrict__ cb_half,
                        float* __restrict__ out_cbv,
                        float* __restrict__ counts) {
  int i = blockIdx.x * blockDim.x + threadIdx.x;
  if (i < NUM_CENTROIDS * DIM) {
    float c = codebook[i];
    cb_half[i] = (_Float16)c;
    out_cbv[i] = c;
  }
  if (i < NUM_CENTROIDS) counts[i] = 0.0f;
}

// ---------------------------------------------------------------------------
// Kernel 1b: cnorm[k] = ||codebook[k]||^2  (one wave32 per centroid)
// ---------------------------------------------------------------------------
__global__ void vq_cnorm(const float* __restrict__ codebook,
                         float* __restrict__ cnorm) {
  int k = blockIdx.x;
  int lane = threadIdx.x;
  const float* row = codebook + (size_t)k * DIM;
  float s = 0.0f;
  for (int j = lane; j < DIM; j += 32) { float v = row[j]; s += v * v; }
  #pragma unroll
  for (int off = 16; off > 0; off >>= 1) s += __shfl_xor(s, off, 32);
  if (lane == 0) cnorm[k] = s;
}

// ---------------------------------------------------------------------------
// Kernel 2: WMMA distance matmul + row-wise argmin + counts.
// Block = 256 threads (8 waves). Each wave owns 16 rows (one WMMA M-tile),
// held in registers as 8 A-fragments for the whole centroid loop.
// Centroid loop: 64 chunks of 32 centroids; each chunk is TDM-DMA'd into a
// double-buffered LDS tile (overlapped with compute), then every wave runs
// 16 x v_wmma_f32_16x16x32_f16 (two N-tiles x 8 K-steps).
// dist_proxy = cnorm[k] - 2*x.c   (||x||^2 is row-constant -> dropped)
// ---------------------------------------------------------------------------
__global__ __launch_bounds__(256) void vq_argmin(
    const float* __restrict__ X,
    const _Float16* __restrict__ cb_half,
    const float* __restrict__ cnorm,
    int* __restrict__ nn,
    float* __restrict__ counts) {
  __shared__ _Float16 ldsX[8 * 16 * DIM];        // 64 KB: 8 waves x 16 rows
  __shared__ _Float16 ldsB[2][CHUNK_HALFS];      // 2 x 16 KB double buffer

  const int tid = threadIdx.x;
  const int lane = tid & 31;
  const int wave = tid >> 5;
  const int half16 = lane >> 4;       // 0: lanes 0-15, 1: lanes 16-31
  const int ntile = lane & 15;        // column index inside 16x16 tile
  const int gRowBase = blockIdx.x * 128 + wave * 16;

  // Stage this wave's 16 rows of X into LDS as f16 (coalesced float4 loads).
  // Each wave reads back only rows it wrote itself -> no barrier needed.
  {
    const float4* xs = (const float4*)(X + (size_t)gRowBase * DIM);
    _Float16* xd = &ldsX[wave * 16 * DIM];
    #pragma unroll
    for (int i = 0; i < 32; ++i) {
      float4 v = xs[i * 32 + lane];
      int o = (i * 32 + lane) * 4;
      xd[o + 0] = (_Float16)v.x; xd[o + 1] = (_Float16)v.y;
      xd[o + 2] = (_Float16)v.z; xd[o + 3] = (_Float16)v.w;
    }
  }

  // A fragments, ISA 16-bit 16x32 layout:
  //  lanes 0-15 : M=lane,    K runs [0..7]  and [16..23]
  //  lanes 16-31: M=lane-16, K runs [8..15] and [24..31]
  v16h afrag[8];
  {
    const _Float16* xrow = &ldsX[(wave * 16 + ntile) * DIM];
    #pragma unroll
    for (int dk = 0; dk < 8; ++dk) {
      union { v16h v; uint4 q[2]; } u;
      int b0 = dk * 32 + half16 * 8;
      u.q[0] = *(const uint4*)(xrow + b0);
      u.q[1] = *(const uint4*)(xrow + b0 + 16);
      afrag[dk] = u.v;
    }
  }

  float bestV[8];
  int   bestI[8];
  #pragma unroll
  for (int r = 0; r < 8; ++r) { bestV[r] = 3.4e38f; bestI[r] = 0; }

#if HAVE_TDM
  // Prologue: kick DMA for chunk 0 into buffer 0 (one wave issues; TENSORcnt).
  if (wave == 0) tdm_load_chunk(cb_half, (unsigned)(size_t)&ldsB[0][0]);
#endif

  for (int kb = 0; kb < NKB; ++kb) {
    const int cur = kb & 1;
#if HAVE_TDM
    if (wave == 0) __builtin_amdgcn_s_wait_tensorcnt(0);  // buf[cur] landed
    __syncthreads();  // all waves: buf[cur] visible, buf[cur^1] reads retired
    if (wave == 0 && kb + 1 < NKB)
      tdm_load_chunk(cb_half + (size_t)(kb + 1) * CHUNK_HALFS,
                     (unsigned)(size_t)&ldsB[cur ^ 1][0]);
#else
    __syncthreads();
    {  // cooperative sync copy fallback: 16 KB via 256 threads x 4 uint4
      const uint4* src = (const uint4*)(cb_half + (size_t)kb * CHUNK_HALFS);
      uint4* dst = (uint4*)&ldsB[cur][0];
      #pragma unroll
      for (int t = 0; t < 4; ++t) dst[tid + t * 256] = src[tid + t * 256];
    }
    if (kb + 1 < NKB)
      __builtin_prefetch(cb_half + (size_t)(kb + 1) * CHUNK_HALFS + tid * 32, 0, 0);
    __syncthreads();
#endif

    // Two 16-centroid N-tiles per chunk: 16 WMMAs per barrier.
    v8f acc0 = {}, acc1 = {};
    #pragma unroll
    for (int dk = 0; dk < 8; ++dk) {
      // B fragment, 16-bit 32x16 layout: column n = lane&15,
      // lanes 0-15 hold K=0..7,16..23, lanes 16-31 hold K=8..15,24..31.
      union { v16h v; uint4 q[2]; } u0, u1;
      const _Float16* b0 = &ldsB[cur][ntile * DIM + dk * 32 + half16 * 8];
      const _Float16* b1 = b0 + 16 * DIM;
      u0.q[0] = *(const uint4*)(b0);
      u0.q[1] = *(const uint4*)(b0 + 16);
      u1.q[0] = *(const uint4*)(b1);
      u1.q[1] = *(const uint4*)(b1 + 16);
      acc0 = __builtin_amdgcn_wmma_f32_16x16x32_f16(
          false, afrag[dk], false, u0.v, (short)0, acc0, false, false);
      acc1 = __builtin_amdgcn_wmma_f32_16x16x32_f16(
          false, afrag[dk], false, u1.v, (short)0, acc1, false, false);
    }

    const int cid0 = kb * KB_TILE + ntile;
    const int cid1 = cid0 + 16;
    const float cn0 = cnorm[cid0];
    const float cn1 = cnorm[cid1];
    #pragma unroll
    for (int r = 0; r < 8; ++r) {
      float d0 = cn0 - 2.0f * acc0[r];
      if (d0 < bestV[r]) { bestV[r] = d0; bestI[r] = cid0; }
      float d1 = cn1 - 2.0f * acc1[r];
      if (d1 < bestV[r]) { bestV[r] = d1; bestI[r] = cid1; }
    }
  }

  // C/D layout: lane holds column n=lane&15; VGPR r is row m=r (+8 for lanes>=16).
  // Min-reduce each row across its 16 lanes (butterfly stays inside the half).
  #pragma unroll
  for (int r = 0; r < 8; ++r) {
    float v = bestV[r]; int i = bestI[r];
    #pragma unroll
    for (int off = 1; off < 16; off <<= 1) {
      float ov = __shfl_xor(v, off, 32);
      int   oi = __shfl_xor(i, off, 32);
      if (ov < v || (ov == v && oi < i)) { v = ov; i = oi; }
    }
    if (ntile == 0) {
      int row = gRowBase + r + half16 * 8;
      nn[row] = i;
      atomicAdd(&counts[i], 1.0f);
    }
  }
}

// ---------------------------------------------------------------------------
// Kernel 3: gather quantized = codebook[nn], loss = 1.25*(q-x)^2.
// ---------------------------------------------------------------------------
__global__ void vq_gather(const float* __restrict__ X,
                          const float* __restrict__ codebook,
                          const int* __restrict__ nn,
                          float* __restrict__ out_q,
                          float* __restrict__ out_loss) {
  int n = blockIdx.x;
  int d = threadIdx.x;
  int idx = nn[n];
  size_t o = (size_t)n * DIM + d;
  float q = codebook[(size_t)idx * DIM + d];
  float x = X[o];
  float diff = q - x;
  out_q[o] = q;                               // straight-through forward value
  out_loss[o] = (1.0f + COMMIT) * diff * diff;
}

// ---------------------------------------------------------------------------
// Kernel 4: nn_idx as float output + EMA count update.
// ---------------------------------------------------------------------------
__global__ void vq_final(const int* __restrict__ nn,
                         const float* __restrict__ counts,
                         const float* __restrict__ old_counts,
                         float* __restrict__ out_nn,
                         float* __restrict__ out_counts) {
  int i = blockIdx.x * blockDim.x + threadIdx.x;
  if (i < N_ROWS) out_nn[i] = (float)nn[i];
  if (i < NUM_CENTROIDS)
    out_counts[i] = EMA_DECAY * old_counts[i] + (1.0f - EMA_DECAY) * counts[i];
}

// ---------------------------------------------------------------------------
extern "C" void kernel_launch(void* const* d_in, const int* in_sizes, int n_in,
                              void* d_out, int out_size, void* d_ws, size_t ws_size,
                              hipStream_t stream) {
  const float* X          = (const float*)d_in[0];  // [32,1024,256]
  const float* codebook   = (const float*)d_in[1];  // [2048,256]
  const float* old_counts = (const float*)d_in[2];  // [2048]

  // Outputs concatenated flat, reference return order (all f32):
  float* out_q    = (float*)d_out;                          // 8,388,608
  float* out_loss = out_q    + (size_t)N_ROWS * DIM;        // 8,388,608
  float* out_nn   = out_loss + (size_t)N_ROWS * DIM;        //    32,768
  float* out_cbv  = out_nn   + N_ROWS;                      //   524,288
  float* out_cnt  = out_cbv  + (size_t)NUM_CENTROIDS * DIM; //     2,048

  // Workspace (~1.2 MB): f16 codebook, cnorm, counts, nn indices.
  char* ws = (char*)d_ws;
  _Float16* cb_half = (_Float16*)ws;  ws += (size_t)NUM_CENTROIDS * DIM * 2;
  float*    cnorm   = (float*)ws;     ws += (size_t)NUM_CENTROIDS * 4;
  float*    counts  = (float*)ws;     ws += (size_t)NUM_CENTROIDS * 4;
  int*      nn      = (int*)ws;

  vq_prep  <<<(NUM_CENTROIDS * DIM + 255) / 256, 256, 0, stream>>>(codebook, cb_half, out_cbv, counts);
  vq_cnorm <<<NUM_CENTROIDS, 32, 0, stream>>>(codebook, cnorm);
  vq_argmin<<<N_ROWS / 128, 256, 0, stream>>>(X, cb_half, cnorm, nn, counts);
  vq_gather<<<N_ROWS, DIM, 0, stream>>>(X, codebook, nn, out_q, out_loss);
  vq_final <<<(N_ROWS + 255) / 256, 256, 0, stream>>>(nn, counts, old_counts, out_nn, out_cnt);
}
